// CodeARmodel_23433341567092
// MI455X (gfx1250) — matrix operand order repaired and checked by reading, not compiled, not run on hardware
//
#include <hip/hip_runtime.h>
#include <stdint.h>
#include <stddef.h>

// ---------------- problem constants ----------------
#define HD   512
#define NC   1024
#define TT   512
#define LB   128
#define BB   64

// persistent recurrent kernel partition
#define NB1  64     // LSTM1 blocks, 8 hidden units each
#define HPB1 8
#define CB1  32     // 4 gates * 8 units
#define NB2  128    // LSTM2 blocks, 4 hidden units each
#define HPB2 4
#define CB2  16     // 4 gates * 4 units
#define NBP  32     // projection blocks, 32 codes each
#define CBP  32
#define NBS  8      // softmax blocks, 8 batch rows each
#define RING 8      // logits ring depth (backpressure)
#define NBLK (NB1 + NB2 + NBP + NBS)

#define USE_TDM 1   // flip to 0 if tensor_load_to_lds inline asm fails to assemble

typedef __bf16 bf16;
typedef __attribute__((ext_vector_type(16))) __bf16 v16bf;
typedef __attribute__((ext_vector_type(8)))  __bf16 bf16x8;
typedef __attribute__((ext_vector_type(8)))  float  f32x8;
typedef __attribute__((ext_vector_type(4)))  unsigned int u32x4;
typedef __attribute__((ext_vector_type(8)))  unsigned int u32x8;

// ---------------- small device helpers ----------------
__device__ __forceinline__ unsigned short f2bfu(float f) {
  union { float f; unsigned u; } v; v.f = f;
  unsigned r = v.u + 0x7FFFu + ((v.u >> 16) & 1u);   // round-to-nearest-even
  return (unsigned short)(r >> 16);
}
__device__ __forceinline__ float bfu2f(unsigned short h) {
  union { unsigned u; float f; } v; v.u = ((unsigned)h) << 16;
  return v.f;
}
__device__ __forceinline__ bf16 f2bf(float f) {
  unsigned short u = f2bfu(f);
  return __builtin_bit_cast(__bf16, u);
}
__device__ __forceinline__ float bf2f(bf16 b) {
  return bfu2f(__builtin_bit_cast(unsigned short, b));
}
__device__ __forceinline__ float sigm(float x) { return 1.f / (1.f + __expf(-x)); }

// deterministic dropout mask (p=0.5, inverted scaling -> {0, 2})
__device__ __forceinline__ float drop_mask(unsigned which, unsigned t, unsigned b, unsigned h) {
  unsigned x = which * 0x9E3779B9u ^ t * 0x85EBCA6Bu ^ b * 0xC2B2AE35u ^ h * 0x27D4EB2Fu;
  x ^= x >> 16; x *= 0x7FEB352Du; x ^= x >> 15; x *= 0x846CA68Bu; x ^= x >> 16;
  return (x & 1u) ? 2.0f : 0.0f;
}

// ---------------- WMMA fragment helpers ----------------
// A fragment (16x32 bf16, M rows across lanes 0-15; lane>=16 handles k+8/k+24 chunks)
__device__ __forceinline__ v16bf load_a(const bf16* __restrict__ A, int lda, int lane, int k0) {
  const bf16* p = A + (size_t)(lane & 15) * lda + k0 + ((lane >> 4) << 3);
  bf16x8 lo = *(const bf16x8*)p;
  bf16x8 hi = *(const bf16x8*)(p + 16);
  v16bf r;
#pragma unroll
  for (int i = 0; i < 8; ++i) { r[i] = lo[i]; r[i + 8] = hi[i]; }
  return r;
}
// B fragment (32x16 bf16, N across lanes 0-15; lane>=16 handles k+8/k+24 chunks) from K-major [K][ldn] bf16
__device__ __forceinline__ v16bf load_b(const bf16* __restrict__ W, int ldn, int lane, int k0, int n0) {
  const bf16* p = W + (size_t)(k0 + ((lane >> 4) << 3)) * ldn + n0 + (lane & 15);
  v16bf r;
#pragma unroll
  for (int i = 0; i < 8; ++i) { r[i] = p[(size_t)i * ldn]; r[i + 8] = p[(size_t)(16 + i) * ldn]; }
  return r;
}
__device__ __forceinline__ f32x8 wmma_bf16(v16bf a, v16bf b, f32x8 c) {
  return __builtin_amdgcn_wmma_f32_16x16x32_bf16(false, a, false, b, (short)0, c, false, false);
}

// ---------------- flags (acquire/release over L2) ----------------
__device__ __forceinline__ void waitge(int* p, int v) {
  if (threadIdx.x == 0) {
    while (__hip_atomic_load(p, __ATOMIC_ACQUIRE, __HIP_MEMORY_SCOPE_AGENT) < v)
      __builtin_amdgcn_s_sleep(2);
  }
  __syncthreads();
}
__device__ __forceinline__ void bump(int* p) {
  __hip_atomic_fetch_add(p, 1, __ATOMIC_RELEASE, __HIP_MEMORY_SCOPE_AGENT);
}

// ---------------- TDM weight staging: global -> LDS ----------------
__device__ __forceinline__ void stage_weights(const bf16* __restrict__ g, bf16* l, unsigned bytes) {
#if USE_TDM
  if (threadIdx.x < 32) {  // one wave issues the tensor DMA (scalar op)
    unsigned long long ga = (unsigned long long)(const void*)g;   // global byte address
    unsigned lad = (unsigned)(unsigned long long)(void*)l;        // LDS offset = addr[31:0]
    unsigned E = bytes >> 1;                                       // elements, data_size = 2B
    u32x4 g0;
    g0[0] = 1u;                                                    // count = 1 valid descriptor
    g0[1] = lad;                                                   // lds_addr
    g0[2] = (unsigned)ga;                                          // global_addr[31:0]
    g0[3] = ((unsigned)(ga >> 32) & 0x01FFFFFFu) | (2u << 30);     // global_addr[56:32] | type=2
    u32x8 g1;
    g1[0] = 1u << 16;                                              // data_size=1 (2 bytes), mask=0
    g1[1] = (E & 0xFFFFu) << 16;                                   // tensor_dim0[15:0]
    g1[2] = (E >> 16) | (1u << 16);                                // tensor_dim0[31:16] | tensor_dim1=1
    g1[3] = (E & 0xFFFFu) << 16;                                   // tile_dim0 (<=16384 here)
    g1[4] = 1u;                                                    // tile_dim1=1, tile_dim2=0
    g1[5] = E;                                                     // tensor_dim0_stride[31:0]
    g1[6] = 0u;                                                    // stride hi / dim1_stride
    g1[7] = 0u;
    asm volatile("tensor_load_to_lds %0, %1" :: "s"(g0), "s"(g1) : "memory");
    __builtin_amdgcn_s_wait_tensorcnt(0);
  }
#else
  for (unsigned i = threadIdx.x * 8u; i < (bytes >> 1); i += 256u * 8u)
    *(bf16x8*)(l + i) = *(const bf16x8*)(g + i);
#endif
  __syncthreads();
}

// ==================================================================
// Small prep kernels
// ==================================================================
__global__ __launch_bounds__(256) void k_cvt(const float* __restrict__ s, unsigned short* __restrict__ d, int n) {
  int i = blockIdx.x * 256 + threadIdx.x;
  if (i < n) d[i] = f2bfu(s[i]);
}
__global__ __launch_bounds__(256) void k_vecadd(const float* __restrict__ a, const float* __restrict__ b,
                                                float* __restrict__ o, int n) {
  int i = blockIdx.x * 256 + threadIdx.x;
  if (i < n) o[i] = a[i] + b[i];
}
// W [N,K] fp32 row-major -> K-major bf16 [K][N]
__global__ __launch_bounds__(256) void k_pack_kn(const float* __restrict__ s, unsigned short* __restrict__ d,
                                                 int N, int K) {
  int i = blockIdx.x * 256 + threadIdx.x;
  if (i < N * K) {
    int n = i % N, k = i / N;
    d[i] = f2bfu(s[(size_t)n * K + k]);
  }
}
// per-block gate-sliced K-major pack: dst[blk][k][c], src [Ntot,512] fp32
__global__ __launch_bounds__(256) void k_pack_sliced(const float* __restrict__ s, unsigned short* __restrict__ d,
                                                     int nblk, int CB, int hper, int group) {
  int i = blockIdx.x * 256 + threadIdx.x;
  if (i < nblk * 512 * CB) {
    int c = i % CB, k = (i / CB) % 512, b = i / (CB * 512);
    int gr = group ? (c / hper) * group + b * hper + (c % hper) : b * CB + c;
    d[i] = f2bfu(s[(size_t)gr * 512 + k]);
  }
}
__global__ __launch_bounds__(256) void k_embed(const int* __restrict__ x, const float* __restrict__ emb,
                                               unsigned short* __restrict__ o) {
  int i = blockIdx.x * 256 + threadIdx.x;         // over B*T*H, row = b*T + t
  int h = i & (HD - 1), row = i >> 9;
  o[i] = f2bfu(emb[(size_t)x[row] * HD + h]);
}
// A1[t][b][h] = ((conds[b][h] + shift) * m1(t,b,h)) as bf16
__global__ __launch_bounds__(256) void k_prep_a1(const unsigned short* __restrict__ XE,
                                                 const float* __restrict__ conds,
                                                 const float* __restrict__ sos,
                                                 unsigned short* __restrict__ A1) {
  int i = blockIdx.x * 256 + threadIdx.x;         // over T*B*H
  int h = i & (HD - 1), bt = i >> 9, b = bt & (BB - 1), t = bt >> 6;
  float xs = (t == 0) ? sos[h] : bfu2f(XE[((size_t)b * TT + (t - 1)) * HD + h]);
  float v = (conds[b * HD + h] + xs) * drop_mask(1u, t, b, h);
  A1[i] = f2bfu(v);
}
__global__ __launch_bounds__(256) void k_initf(int* __restrict__ flags) {
  int i = blockIdx.x * 256 + threadIdx.x;
  if (i < 2562) {
    int v = 0;
    if (i == 0) v = NB1;        // flagA[0]: h1 slot 0 (zeros) valid
    if (i == 1025) v = NB2;     // flagC[0]: h2 slot 0 (zeros) valid
    flags[i] = v;
  }
}
__global__ __launch_bounds__(256) void k_zero16(unsigned short* __restrict__ p, int n) {
  int i = blockIdx.x * 256 + threadIdx.x;
  if (i < n) p[i] = 0;
}

// ==================================================================
// Generic tiled WMMA GEMM: C = act(A[M,K]bf16 @ Wkn[K,N]bf16 + bias)
// 64x64 block tile, BK=32, 8 waves each computing two 16x16 subtiles.
// ==================================================================
__global__ __launch_bounds__(256) void k_gemm(const bf16* __restrict__ A, const bf16* __restrict__ W,
                                              const float* __restrict__ bias, void* __restrict__ Cout,
                                              int M, int N, int K, int relu, int out_bf16) {
  __shared__ __align__(64) bf16 As[64][32];
  __shared__ __align__(64) bf16 Bs[32][64];
  const int n0 = blockIdx.x * 64, m0 = blockIdx.y * 64;
  const int tid = threadIdx.x, wave = tid >> 5, lane = tid & 31;
  const int miB = (wave >> 2) * 2, ni = wave & 3;
  f32x8 acc0 = {0.f,0.f,0.f,0.f,0.f,0.f,0.f,0.f};
  f32x8 acc1 = {0.f,0.f,0.f,0.f,0.f,0.f,0.f,0.f};
  for (int k0 = 0; k0 < K; k0 += 32) {
    { int r = tid >> 2, c = (tid & 3) * 8;
      *(bf16x8*)&As[r][c] = *(const bf16x8*)(A + (size_t)(m0 + r) * K + k0 + c); }
    { int r = tid >> 3, c = (tid & 7) * 8;
      *(bf16x8*)&Bs[r][c] = *(const bf16x8*)(W + (size_t)(k0 + r) * N + n0 + c); }
    __syncthreads();
    v16bf bfrag = load_b(&Bs[0][0], 64, lane, 0, ni * 16);
    v16bf a0 = load_a(&As[0][0] + (size_t)miB * 16 * 32, 32, lane, 0);
    v16bf a1 = load_a(&As[0][0] + (size_t)(miB + 1) * 16 * 32, 32, lane, 0);
    acc0 = wmma_bf16(a0, bfrag, acc0);
    acc1 = wmma_bf16(a1, bfrag, acc1);
    __syncthreads();
  }
  const int cg = n0 + ni * 16 + (lane & 15);
  const float bv = bias ? bias[cg] : 0.f;
  const int r0 = m0 + miB * 16 + ((lane >> 4) << 3);
#pragma unroll
  for (int r = 0; r < 8; ++r) {
    float v0 = acc0[r] + bv, v1 = acc1[r] + bv;
    if (relu) { v0 = fmaxf(v0, 0.f); v1 = fmaxf(v1, 0.f); }
    size_t i0 = (size_t)(r0 + r) * N + cg, i1 = (size_t)(r0 + 16 + r) * N + cg;
    if (out_bf16) { ((unsigned short*)Cout)[i0] = f2bfu(v0); ((unsigned short*)Cout)[i1] = f2bfu(v1); }
    else          { ((float*)Cout)[i0] = v0;                 ((float*)Cout)[i1] = v1; }
  }
}

// ==================================================================
// Persistent recurrent kernel: LSTM1 | LSTM2 | PROJ | SOFTMAX pipeline
// ==================================================================
__global__ __launch_bounds__(256) void k_recur(
    const bf16* __restrict__ w1s, const bf16* __restrict__ w2is,
    const bf16* __restrict__ w2hs, const bf16* __restrict__ wps,
    const bf16* __restrict__ pre1, const float* __restrict__ b2sum,
    const float* __restrict__ proj_b,
    bf16* __restrict__ h1, bf16* __restrict__ h1m, bf16* __restrict__ h2,
    float* __restrict__ lring, int* __restrict__ flags, float* __restrict__ out) {
  __shared__ __align__(64) unsigned char smem[49152];
  int* fA = flags;            // [513] writers of h1 slot t
  int* fB = flags + 513;      // [512] writers of h1m[t]
  int* fC = flags + 1025;     // [513] writers of h2 slot t
  int* fP = flags + 1538;     // [512] projection done for step t
  int* fS = flags + 2050;     // [512] softmax consumed step t
  const int tid = threadIdx.x, wave = tid >> 5, lane = tid & 31;
  const int blk = blockIdx.x;

  if (blk < NB1) {
    // -------- LSTM1: gates = pre1[t] + h1[t] @ Whh1-slice --------
    const int w = blk;
    bf16* Wt = (bf16*)smem;                            // [512][32] 32KB
    float* gates = (float*)(smem + 32768);             // [64][32]  8KB
    float* cst = (float*)(smem + 40960);               // [64][8]   2KB
    stage_weights(w1s + (size_t)w * 512 * CB1, Wt, 512 * CB1 * 2);
    for (int i = tid; i < BB * HPB1; i += 256) cst[i] = 0.f;
    __builtin_amdgcn_s_cluster_barrier();
    __syncthreads();
    for (int t = 0; t < TT; ++t) {
      if (t + 1 < TT) __builtin_prefetch(pre1 + ((size_t)(t + 1) * BB) * 2048, 0, 1);
      waitge(fA + t, NB1);
      for (int i = tid; i < BB * CB1; i += 256) {
        int b = i >> 5, c = i & 31;
        int gr = (c >> 3) * 512 + w * HPB1 + (c & 7);
        gates[i] = bf2f(pre1[((size_t)t * BB + b) * 2048 + gr]);
      }
      __syncthreads();
      const bf16* Hp = h1 + (size_t)t * BB * HD;
      const int mi = wave >> 1, ni = wave & 1;
      f32x8 acc = {0.f,0.f,0.f,0.f,0.f,0.f,0.f,0.f};
#pragma unroll
      for (int k0 = 0; k0 < HD; k0 += 32)
        acc = wmma_bf16(load_a(Hp + (size_t)mi * 16 * HD, HD, lane, k0),
                        load_b(Wt, CB1, lane, k0, ni * 16), acc);
      { int r0 = mi * 16 + ((lane >> 4) << 3), c0 = ni * 16 + (lane & 15);
#pragma unroll
        for (int r = 0; r < 8; ++r) gates[(r0 + r) * CB1 + c0] += acc[r]; }
      __syncthreads();
      if (tid < BB * HPB1) {
        int b = tid >> 3, j = tid & 7;
        float gi = sigm(gates[b * CB1 + j]);
        float gf = sigm(gates[b * CB1 + 8 + j]);
        float gg = tanhf(gates[b * CB1 + 16 + j]);
        float go = sigm(gates[b * CB1 + 24 + j]);
        float c = gf * cst[tid] + gi * gg;
        cst[tid] = c;
        float hn = go * tanhf(c);
        int hidx = w * HPB1 + j;
        h1[((size_t)(t + 1) * BB + b) * HD + hidx] = f2bf(hn);
        h1m[((size_t)t * BB + b) * HD + hidx] = f2bf(hn * drop_mask(2u, t, b, hidx));
      }
      __syncthreads();
      if (tid == 0) { __threadfence(); bump(fA + t + 1); bump(fB + t); }
    }
  } else if (blk < NB1 + NB2) {
    // -------- LSTM2: gates = b2 + h1m[t]@Wih2 + h2[t]@Whh2 --------
    const int w = blk - NB1;
    bf16* Wi = (bf16*)smem;                            // [512][16] 16KB
    bf16* Wh = (bf16*)(smem + 16384);                  // [512][16] 16KB
    float* gates = (float*)(smem + 32768);             // [64][16]  4KB
    float* cst = (float*)(smem + 36864);               // [64][4]   1KB
    stage_weights(w2is + (size_t)w * 512 * CB2, Wi, 512 * CB2 * 2);
    stage_weights(w2hs + (size_t)w * 512 * CB2, Wh, 512 * CB2 * 2);
    for (int i = tid; i < BB * HPB2; i += 256) cst[i] = 0.f;
    __builtin_amdgcn_s_cluster_barrier();
    __syncthreads();
    for (int t = 0; t < TT; ++t) {
      waitge(fB + t, NB1);
      waitge(fC + t, NB2);
      for (int i = tid; i < BB * CB2; i += 256) {
        int c = i & 15;
        gates[i] = b2sum[(c >> 2) * 512 + w * HPB2 + (c & 3)];
      }
      __syncthreads();
      const bf16* A = (wave < 4) ? (h1m + (size_t)t * BB * HD) : (h2 + (size_t)t * BB * HD);
      const bf16* Wm = (wave < 4) ? Wi : Wh;
      const int mi = wave & 3;
      f32x8 acc = {0.f,0.f,0.f,0.f,0.f,0.f,0.f,0.f};
#pragma unroll
      for (int k0 = 0; k0 < HD; k0 += 32)
        acc = wmma_bf16(load_a(A + (size_t)mi * 16 * HD, HD, lane, k0),
                        load_b(Wm, CB2, lane, k0, 0), acc);
      { int r0 = mi * 16 + ((lane >> 4) << 3), c0 = lane & 15;
#pragma unroll
        for (int r = 0; r < 8; ++r) atomicAdd(&gates[(r0 + r) * CB2 + c0], acc[r]); }
      __syncthreads();
      if (tid < BB * HPB2) {
        int b = tid >> 2, j = tid & 3;
        float gi = sigm(gates[b * CB2 + j]);
        float gf = sigm(gates[b * CB2 + 4 + j]);
        float gg = tanhf(gates[b * CB2 + 8 + j]);
        float go = sigm(gates[b * CB2 + 12 + j]);
        float c = gf * cst[tid] + gi * gg;
        cst[tid] = c;
        float hn = go * tanhf(c);
        h2[((size_t)(t + 1) * BB + b) * HD + w * HPB2 + j] = f2bf(hn);
      }
      __syncthreads();
      if (tid == 0) { __threadfence(); bump(fC + t + 1); }
    }
  } else if (blk < NB1 + NB2 + NBP) {
    // -------- PROJ: logits slice = h2[t+1] @ projW-slice + b --------
    const int p = blk - NB1 - NB2;
    bf16* Wt = (bf16*)smem;                            // [512][32] 32KB
    stage_weights(wps + (size_t)p * 512 * CBP, Wt, 512 * CBP * 2);
    __builtin_amdgcn_s_cluster_barrier();
    __syncthreads();
    for (int t = 0; t < TT; ++t) {
      waitge(fC + t + 1, NB2);
      if (t >= RING) waitge(fS + t - RING, NBS);       // ring slot free
      float* lg = lring + (size_t)(t & (RING - 1)) * BB * NC;
      const bf16* Hp = h2 + (size_t)(t + 1) * BB * HD;
      const int mi = wave >> 1, ni = wave & 1;
      f32x8 acc = {0.f,0.f,0.f,0.f,0.f,0.f,0.f,0.f};
#pragma unroll
      for (int k0 = 0; k0 < HD; k0 += 32)
        acc = wmma_bf16(load_a(Hp + (size_t)mi * 16 * HD, HD, lane, k0),
                        load_b(Wt, CBP, lane, k0, ni * 16), acc);
      const int r0 = mi * 16 + ((lane >> 4) << 3);
      const int cg = p * CBP + ni * 16 + (lane & 15);
      const float bv = proj_b[cg];
#pragma unroll
      for (int r = 0; r < 8; ++r) lg[(size_t)(r0 + r) * NC + cg] = acc[r] + bv;
      __syncthreads();
      if (tid == 0) { __threadfence(); bump(fP + t); }
    }
  } else {
    // -------- SOFTMAX: log_softmax over 1024 codes, 8 rows/block --------
    const int s = blk - NB1 - NB2 - NBP;
    float* red = (float*)smem;                         // [256]
    for (int t = 0; t < TT; ++t) {
      waitge(fP + t, NBP);
      const float* lg = lring + (size_t)(t & (RING - 1)) * BB * NC;
      for (int rb = 0; rb < 8; ++rb) {
        const int b = s * 8 + rb;
        const float* row = lg + (size_t)b * NC;
        float m = -3.4e38f;
        for (int j = tid; j < NC; j += 256) m = fmaxf(m, row[j]);
        red[tid] = m; __syncthreads();
        for (int st = 128; st > 0; st >>= 1) { if (tid < st) red[tid] = fmaxf(red[tid], red[tid + st]); __syncthreads(); }
        const float M = red[0]; __syncthreads();
        float sum = 0.f;
        for (int j = tid; j < NC; j += 256) sum += __expf(row[j] - M);
        red[tid] = sum; __syncthreads();
        for (int st = 128; st > 0; st >>= 1) { if (tid < st) red[tid] += red[tid + st]; __syncthreads(); }
        const float L = M + __logf(red[0]); __syncthreads();
        float* op = out + ((size_t)b * TT + t) * NC;
        for (int j = tid; j < NC; j += 256) op[j] = row[j] - L;
      }
      __syncthreads();
      if (tid == 0) { __threadfence(); bump(fS + t); }
    }
  }
}

// ==================================================================
// Host launcher
// ==================================================================
extern "C" void kernel_launch(void* const* d_in, const int* in_sizes, int n_in,
                              void* d_out, int out_size, void* d_ws, size_t ws_size,
                              hipStream_t stream) {
  (void)in_sizes; (void)n_in; (void)out_size; (void)ws_size;
  const int*   x      = (const int*)d_in[0];
  const float* labels = (const float*)d_in[1];
  const float* emb    = (const float*)d_in[2];
  const float* xl_w1  = (const float*)d_in[3];
  const float* xl_b1  = (const float*)d_in[4];
  const float* xl_w2  = (const float*)d_in[5];
  const float* xl_b2  = (const float*)d_in[6];
  const float* xl_w3  = (const float*)d_in[7];
  const float* ll_w1  = (const float*)d_in[8];
  const float* ll_b1  = (const float*)d_in[9];
  const float* ll_w2  = (const float*)d_in[10];
  const float* ll_b2  = (const float*)d_in[11];
  const float* ll_w3  = (const float*)d_in[12];
  const float* l1_wih = (const float*)d_in[13];
  const float* l1_whh = (const float*)d_in[14];
  const float* l1_bih = (const float*)d_in[15];
  const float* l1_bhh = (const float*)d_in[16];
  const float* l2_wih = (const float*)d_in[17];
  const float* l2_whh = (const float*)d_in[18];
  const float* l2_bih = (const float*)d_in[19];
  const float* l2_bhh = (const float*)d_in[20];
  const float* proj_w = (const float*)d_in[21];
  const float* proj_b = (const float*)d_in[22];
  const float* sos    = (const float*)d_in[23];

  size_t off = 0;
  auto take = [&](size_t bytes) -> char* {
    char* p = (char*)d_ws + off;
    off += (bytes + 255) & ~(size_t)255;
    return p;
  };
  const size_t NBT = (size_t)BB * TT;               // 32768 rows
  unsigned short* bufA   = (unsigned short*)take(NBT * HD * 2);
  unsigned short* bufB   = (unsigned short*)take(NBT * HD * 2);
  unsigned short* pre1   = (unsigned short*)take(NBT * 2048 * 2);
  unsigned short* wkn_x1 = (unsigned short*)take((size_t)HD * HD * 2);
  unsigned short* wkn_x2 = (unsigned short*)take((size_t)HD * HD * 2);
  unsigned short* wkn_x3 = (unsigned short*)take((size_t)HD * HD * 2);
  unsigned short* wkn_l1 = (unsigned short*)take((size_t)LB * HD * 2);
  unsigned short* wkn_l2 = (unsigned short*)take((size_t)HD * HD * 2);
  unsigned short* wkn_l3 = (unsigned short*)take((size_t)HD * HD * 2);
  unsigned short* wkn_w1 = (unsigned short*)take((size_t)HD * 2048 * 2);
  unsigned short* w1s    = (unsigned short*)take((size_t)NB1 * 512 * CB1 * 2);
  unsigned short* w2is   = (unsigned short*)take((size_t)NB2 * 512 * CB2 * 2);
  unsigned short* w2hs   = (unsigned short*)take((size_t)NB2 * 512 * CB2 * 2);
  unsigned short* wps    = (unsigned short*)take((size_t)NBP * 512 * CBP * 2);
  float* b1sum           = (float*)take(2048 * 4);
  float* b2sum           = (float*)take(2048 * 4);
  unsigned short* lab_bf = (unsigned short*)take((size_t)BB * LB * 2);
  unsigned short* lg1    = (unsigned short*)take((size_t)BB * HD * 2);
  unsigned short* lg2    = (unsigned short*)take((size_t)BB * HD * 2);
  float* conds           = (float*)take((size_t)BB * HD * 4);
  unsigned short* h1     = (unsigned short*)take((size_t)(TT + 1) * BB * HD * 2);
  unsigned short* h1m    = (unsigned short*)take((size_t)TT * BB * HD * 2);
  unsigned short* h2     = (unsigned short*)take((size_t)(TT + 1) * BB * HD * 2);
  float* lring           = (float*)take((size_t)RING * BB * NC * 4);
  int* flags             = (int*)take(2562 * 4);

  // ---- state / flag init, weight packing ----
  k_initf<<<11, 256, 0, stream>>>(flags);
  k_zero16<<<128, 256, 0, stream>>>(h1, BB * HD);
  k_zero16<<<128, 256, 0, stream>>>(h2, BB * HD);
  k_vecadd<<<8, 256, 0, stream>>>(l1_bih, l1_bhh, b1sum, 2048);
  k_vecadd<<<8, 256, 0, stream>>>(l2_bih, l2_bhh, b2sum, 2048);
  k_pack_kn<<<1024, 256, 0, stream>>>(xl_w1, wkn_x1, HD, HD);
  k_pack_kn<<<1024, 256, 0, stream>>>(xl_w2, wkn_x2, HD, HD);
  k_pack_kn<<<1024, 256, 0, stream>>>(xl_w3, wkn_x3, HD, HD);
  k_pack_kn<<<256, 256, 0, stream>>>(ll_w1, wkn_l1, HD, LB);
  k_pack_kn<<<1024, 256, 0, stream>>>(ll_w2, wkn_l2, HD, HD);
  k_pack_kn<<<1024, 256, 0, stream>>>(ll_w3, wkn_l3, HD, HD);
  k_pack_kn<<<4096, 256, 0, stream>>>(l1_wih, wkn_w1, 2048, HD);
  k_pack_sliced<<<4096, 256, 0, stream>>>(l1_whh, w1s, NB1, CB1, HPB1, 512);
  k_pack_sliced<<<4096, 256, 0, stream>>>(l2_wih, w2is, NB2, CB2, HPB2, 512);
  k_pack_sliced<<<4096, 256, 0, stream>>>(l2_whh, w2hs, NB2, CB2, HPB2, 512);
  k_pack_sliced<<<2048, 256, 0, stream>>>(proj_w, wps, NBP, CBP, 0, 0);

  // ---- parallel prefix: embedding + x-MLP, label-MLP, pre1 GEMM ----
  k_embed<<<65536, 256, 0, stream>>>(x, emb, bufA);
  k_gemm<<<dim3(8, 512), 256, 0, stream>>>((const bf16*)bufA, (const bf16*)wkn_x1, xl_b1, bufB, 32768, HD, HD, 1, 1);
  k_gemm<<<dim3(8, 512), 256, 0, stream>>>((const bf16*)bufB, (const bf16*)wkn_x2, xl_b2, bufA, 32768, HD, HD, 1, 1);
  k_gemm<<<dim3(8, 512), 256, 0, stream>>>((const bf16*)bufA, (const bf16*)wkn_x3, nullptr, bufB, 32768, HD, HD, 0, 1);
  k_cvt<<<32, 256, 0, stream>>>(labels, lab_bf, BB * LB);
  k_gemm<<<dim3(8, 1), 256, 0, stream>>>((const bf16*)lab_bf, (const bf16*)wkn_l1, ll_b1, lg1, BB, HD, LB, 1, 1);
  k_gemm<<<dim3(8, 1), 256, 0, stream>>>((const bf16*)lg1, (const bf16*)wkn_l2, ll_b2, lg2, BB, HD, HD, 1, 1);
  k_gemm<<<dim3(8, 1), 256, 0, stream>>>((const bf16*)lg2, (const bf16*)wkn_l3, nullptr, conds, BB, HD, HD, 0, 0);
  k_prep_a1<<<65536, 256, 0, stream>>>(bufB, conds, sos, bufA);
  k_gemm<<<dim3(32, 512), 256, 0, stream>>>((const bf16*)bufA, (const bf16*)wkn_w1, b1sum, pre1, 32768, 2048, HD, 0, 1);

  // ---- persistent recurrent pipeline ----
  k_recur<<<NBLK, 256, 0, stream>>>((const bf16*)w1s, (const bf16*)w2is, (const bf16*)w2hs,
                                    (const bf16*)wps, (const bf16*)pre1, b2sum, proj_b,
                                    (bf16*)h1, (bf16*)h1m, (bf16*)h2,
                                    lring, flags, (float*)d_out);
}
